// VectorQuantizer_35983236005999
// MI455X (gfx1250) — matrix operand order repaired and checked by reading, not compiled, not run on hardware
//
#include <hip/hip_runtime.h>

#define EMBED_NUM 8192
#define EMBED_DIM 256
#define NB 16
#define NC 256
#define NH 32
#define NW 32
#define HW (NH * NW)          // 1024
#define NTOK (NB * HW)        // 16384

typedef __attribute__((ext_vector_type(16))) __bf16 v16bf;
typedef __attribute__((ext_vector_type(8)))  __bf16 v8bf;
typedef __attribute__((ext_vector_type(8)))  float  v8f;
typedef __attribute__((ext_vector_type(4)))  int    vi4;
typedef __attribute__((address_space(1))) vi4 gvi4;   // global int4
typedef __attribute__((address_space(3))) vi4 lvi4;   // LDS int4

// ---------------------------------------------------------------------------
// Kernel 1: (B,C,H,W) f32 -> x_bf16 (N=16384 x 256) row-major, LDS-tiled
// transpose so both global reads and writes are coalesced.
// ---------------------------------------------------------------------------
__global__ __launch_bounds__(256) void k_prep_x(const float* __restrict__ latent,
                                                __bf16* __restrict__ xb) {
    __shared__ float tile[32][33];
    const int hw0 = blockIdx.x * 32;
    const int c0  = blockIdx.y * 32;
    const int b   = blockIdx.z;
    const float* src = latent + (size_t)b * NC * HW;
    const int tx = threadIdx.x, ty = threadIdx.y;
    #pragma unroll
    for (int i = ty; i < 32; i += 8)
        tile[i][tx] = src[(size_t)(c0 + i) * HW + hw0 + tx];
    __syncthreads();
    #pragma unroll
    for (int i = ty; i < 32; i += 8) {
        const int n = b * HW + hw0 + i;
        xb[(size_t)n * EMBED_DIM + c0 + tx] = (__bf16)tile[tx][i];
    }
}

// ---------------------------------------------------------------------------
// Kernel 2: codebook f32 -> bf16, plus cb2[k] = sum_d codebook[k][d]^2 (f32).
// ---------------------------------------------------------------------------
__global__ __launch_bounds__(256) void k_prep_cb(const float* __restrict__ cb,
                                                 __bf16* __restrict__ cbb,
                                                 float* __restrict__ cb2) {
    const int wave = threadIdx.x >> 5, lane = threadIdx.x & 31;
    const int row  = blockIdx.x * 8 + wave;
    const float* src = cb + (size_t)row * EMBED_DIM;
    __bf16* dst = cbb + (size_t)row * EMBED_DIM;
    float s = 0.f;
    #pragma unroll
    for (int i = 0; i < 8; ++i) {
        const float v = src[lane + i * 32];
        dst[lane + i * 32] = (__bf16)v;
        s += v * v;
    }
    #pragma unroll
    for (int m = 16; m > 0; m >>= 1) s += __shfl_xor(s, m, 32);
    if (lane == 0) cb2[row] = s;
}

// ---------------------------------------------------------------------------
// Kernel 3: WMMA distance GEMM + partial argmin over a K-chunk.
// Grid (128, KSPLIT): x = token block (128 tokens, 8 waves x 16), y = K-chunk
// of 2048 codes. Each wave keeps A (16 tokens x 256 d) in VGPRs; the codebook
// chunk is staged into double-buffered LDS via CDNA5 async global->LDS copies.
// Two 16-code tiles are computed as explicit independent wmma chains so B
// fragment ds_loads pipeline ahead of the matrix ops.
// ---------------------------------------------------------------------------
#define KSPLIT 4
#define STAGE_CODES 32
#define LDS_ROW 272   // bf16 elems per staged code row (544 B = 17*32, 32B-aligned)

#if __has_builtin(__builtin_amdgcn_global_load_async_to_lds_b128)
#define HAVE_ASYNC_LDS 1
#else
#define HAVE_ASYNC_LDS 0
#endif

__global__ __launch_bounds__(256) void k_argmin(const __bf16* __restrict__ xb,
                                                const __bf16* __restrict__ cbb,
                                                const float* __restrict__ cb2,
                                                float* __restrict__ pmin,
                                                int* __restrict__ pidx) {
    __shared__ __align__(32) __bf16 lbuf[2][STAGE_CODES * LDS_ROW];
    __shared__ float lcb2[2][STAGE_CODES];

    const int tid  = threadIdx.x;
    const int wave = tid >> 5;
    const int lane = tid & 31;
    const int half = lane >> 4;     // 0: lanes 0-15, 1: lanes 16-31
    const int l15  = lane & 15;
    const int tok0 = (blockIdx.x * 8 + wave) * 16;
    const int kch  = blockIdx.y;

    // A fragments: row M = lane&15; lanes 0-15 hold K={0..15}+32c,
    // lanes 16-31 hold K={16..31}+32c  (contiguous k order, matches B).
    v16bf afrag[8];
    {
        const __bf16* xrow = xb + (size_t)(tok0 + l15) * EMBED_DIM;
        #pragma unroll
        for (int c = 0; c < 8; ++c) {
            const int d0 = 32 * c + half * 16;
            v8bf lo = *(const v8bf*)(xrow + d0);
            v8bf hi = *(const v8bf*)(xrow + d0 + 8);
            afrag[c] = __builtin_shufflevector(lo, hi,
                0, 1, 2, 3, 4, 5, 6, 7, 8, 9, 10, 11, 12, 13, 14, 15);
        }
    }

    float vmin[8];
    int   vidx[8];
    #pragma unroll
    for (int j = 0; j < 8; ++j) { vmin[j] = 3.4e38f; vidx[j] = 0; }

    // Stage 32 codes (32 x 512 B = 16 KB) into LDS buffer bufsel.
    auto stage_load = [&](int s, int bufsel) {
        const __bf16* g = cbb + (size_t)s * STAGE_CODES * EMBED_DIM;
        #pragma unroll
        for (int u = 0; u < 4; ++u) {                 // 32 rows * 32 16B-units
            const int unit = tid + u * 256;
            const int q    = unit >> 5;               // code row in stage
            const int off  = (unit & 31) * 8;         // bf16 offset
#if HAVE_ASYNC_LDS
            __builtin_amdgcn_global_load_async_to_lds_b128(
                (gvi4*)(g + (size_t)q * EMBED_DIM + off),
                (lvi4*)(&lbuf[bufsel][q * LDS_ROW + off]),
                0, 0);
#else
            *(v8bf*)(&lbuf[bufsel][q * LDS_ROW + off]) =
                *(const v8bf*)(g + (size_t)q * EMBED_DIM + off);
#endif
        }
        if (tid < STAGE_CODES) lcb2[bufsel][tid] = cb2[s * STAGE_CODES + tid];
    };

    auto wait_async = [&]() {
#if HAVE_ASYNC_LDS
#if __has_builtin(__builtin_amdgcn_s_wait_asynccnt)
        __builtin_amdgcn_s_wait_asynccnt(0);
#else
        asm volatile("s_wait_asynccnt 0x0" ::: "memory");
#endif
#endif
    };

    const int NSTAGE = EMBED_NUM / (KSPLIT * STAGE_CODES);   // 64 stages per chunk
    const int s0 = kch * NSTAGE;

    stage_load(s0, 0);
    wait_async();
    __syncthreads();

    for (int s = 0; s < NSTAGE; ++s) {
        const int cur = s & 1;
        if (s + 1 < NSTAGE) stage_load(s0 + s + 1, cur ^ 1);

        // Two independent wmma chains (tiles 0 and 1) so ds_loads run ahead.
        v8f acc0 = {}, acc1 = {};
        const __bf16* b0 = &lbuf[cur][(l15)      * LDS_ROW + half * 16];
        const __bf16* b1 = &lbuf[cur][(16 + l15) * LDS_ROW + half * 16];
        #pragma unroll
        for (int c = 0; c < 8; ++c) {
            v16bf f0 = *(const v16bf*)(b0 + 32 * c);
            v16bf f1 = *(const v16bf*)(b1 + 32 * c);
            acc0 = __builtin_amdgcn_wmma_f32_16x16x32_bf16(
                false, afrag[c], false, f0, (short)0, acc0, false, false);
            acc1 = __builtin_amdgcn_wmma_f32_16x16x32_bf16(
                false, afrag[c], false, f1, (short)0, acc1, false, false);
        }

        const int   base  = (s0 + s) * STAGE_CODES;
        const int   code0 = base + l15;
        const int   code1 = base + 16 + l15;
        const float c20   = lcb2[cur][l15];
        const float c21   = lcb2[cur][16 + l15];
        #pragma unroll
        for (int j = 0; j < 8; ++j) {
            const float sc0 = c20 - 2.0f * acc0[j];
            if (sc0 < vmin[j]) { vmin[j] = sc0; vidx[j] = code0; }
            const float sc1 = c21 - 2.0f * acc1[j];
            if (sc1 < vmin[j]) { vmin[j] = sc1; vidx[j] = code1; }
        }
        wait_async();   // our async loads into the other buffer have landed
        __syncthreads();
    }

    // C/D layout: VGPR j, lanes 0-15 -> row M=j; lanes 16-31 -> row M=j+8.
    // The 16 code-candidates of a row sit across the 16 lanes of one half.
    #pragma unroll
    for (int j = 0; j < 8; ++j) {
        float m = vmin[j]; int ix = vidx[j];
        #pragma unroll
        for (int msk = 8; msk > 0; msk >>= 1) {
            const float om = __shfl_xor(m, msk, 32);
            const int   oi = __shfl_xor(ix, msk, 32);
            if (om < m || (om == m && oi < ix)) { m = om; ix = oi; }
        }
        if (l15 == 0) {
            const int tok = tok0 + j + half * 8;
            pmin[kch * NTOK + tok] = m;
            pidx[kch * NTOK + tok] = ix;
        }
    }
}

// ---------------------------------------------------------------------------
// Kernel 3b: combine KSPLIT partial argmins per token (ascending-k order,
// strict < keeps the first/lowest index on ties, matching jnp.argmin).
// ---------------------------------------------------------------------------
__global__ __launch_bounds__(256) void k_comb(const float* __restrict__ pmin,
                                              const int* __restrict__ pidx,
                                              int* __restrict__ widx) {
    const int n = blockIdx.x * 256 + threadIdx.x;
    float m = pmin[n];
    int  ix = pidx[n];
    #pragma unroll
    for (int q = 1; q < KSPLIT; ++q) {
        const float om = pmin[q * NTOK + n];
        const int   oi = pidx[q * NTOK + n];
        if (om < m) { m = om; ix = oi; }
    }
    widx[n] = ix;
}

// ---------------------------------------------------------------------------
// Kernel 4: gather xq = codebook[idx] (exact f32), write quantized output in
// (B,C,H,W) order (coalesced writes), accumulate per-block sum of (xq-x)^2.
// ---------------------------------------------------------------------------
__global__ __launch_bounds__(256) void k_gather(const float* __restrict__ latent,
                                                const float* __restrict__ cb,
                                                const int* __restrict__ widx,
                                                float* __restrict__ outq,
                                                float* __restrict__ partial) {
    __shared__ float red[256];
    const size_t o = (size_t)blockIdx.x * 256 + threadIdx.x;  // < 4194304
    const int b  = (int)(o >> 18);        // C*HW = 262144
    const int r  = (int)(o & 262143);
    const int c  = r >> 10;               // HW = 1024
    const int hw = r & 1023;
    const int ix = widx[b * HW + hw];
    const float xq = cb[(size_t)ix * EMBED_DIM + c];
    const float x  = latent[o];
    outq[o] = xq;
    const float d = xq - x;
    red[threadIdx.x] = d * d;
    __syncthreads();
    #pragma unroll
    for (int st = 128; st > 0; st >>= 1) {
        if (threadIdx.x < st) red[threadIdx.x] += red[threadIdx.x + st];
        __syncthreads();
    }
    if (threadIdx.x == 0) partial[blockIdx.x] = red[0];
}

// ---------------------------------------------------------------------------
// Kernel 5: deterministic final loss reduction (fixed order, single block).
// codebook_loss == commitment_loss numerically in forward.
// ---------------------------------------------------------------------------
__global__ __launch_bounds__(256) void k_loss(const float* __restrict__ partial,
                                              float* __restrict__ out) {
    __shared__ float red[256];
    float s = 0.f;
    for (int i = threadIdx.x; i < 16384; i += 256) s += partial[i];
    red[threadIdx.x] = s;
    __syncthreads();
    #pragma unroll
    for (int st = 128; st > 0; st >>= 1) {
        if (threadIdx.x < st) red[threadIdx.x] += red[threadIdx.x + st];
        __syncthreads();
    }
    if (threadIdx.x == 0) {
        const float mean = red[0] / (float)((size_t)NTOK * EMBED_DIM);
        out[(size_t)NTOK * EMBED_DIM + 0] = mean;   // codebook_loss
        out[(size_t)NTOK * EMBED_DIM + 1] = mean;   // commitment_loss
    }
}

// ---------------------------------------------------------------------------
extern "C" void kernel_launch(void* const* d_in, const int* in_sizes, int n_in,
                              void* d_out, int out_size, void* d_ws, size_t ws_size,
                              hipStream_t stream) {
    (void)in_sizes; (void)n_in; (void)out_size; (void)ws_size;
    const float* latent   = (const float*)d_in[0];
    const float* codebook = (const float*)d_in[1];
    float* out = (float*)d_out;

    char* ws = (char*)d_ws;
    __bf16* xb  = (__bf16*)ws;                          // 8 MB
    __bf16* cbb = (__bf16*)(ws + 8388608);              // 4 MB
    float*  cb2 = (float*)(ws + 12582912);              // 32 KB
    int*    wix = (int*)(ws + 12615680);                // 64 KB
    float*  prt = (float*)(ws + 12681216);              // 64 KB
    float*  pmn = (float*)(ws + 12746752);              // 256 KB (KSPLIT*NTOK)
    int*    pix = (int*)(ws + 13008896);                // 256 KB

    k_prep_x <<<dim3(HW / 32, NC / 32, NB), dim3(32, 8), 0, stream>>>(latent, xb);
    k_prep_cb<<<EMBED_NUM / 8, 256, 0, stream>>>(codebook, cbb, cb2);
    k_argmin <<<dim3(NTOK / 128, KSPLIT), 256, 0, stream>>>(xb, cbb, cb2, pmn, pix);
    k_comb   <<<NTOK / 256, 256, 0, stream>>>(pmn, pix, wix);
    k_gather <<<(NTOK * EMBED_DIM) / 256, 256, 0, stream>>>(latent, codebook, wix, out, prt);
    k_loss   <<<1, 256, 0, stream>>>(prt, out);
}
